// GATModel_84559316124080
// MI455X (gfx1250) — matrix operand order repaired and checked by reading, not compiled, not run on hardware
//
#include <hip/hip_runtime.h>

#define NNODE 50000
#define NEDGE 400000
#define FIN   128
#define HEADS 4
#define DHID  128
#define HD    512      // HEADS*DHID
#define NC1   1024     // fused z||res output cols, layer 1
#define NC2   32       // fused z||res output cols, layer 2 (3 used + pad, 3 used + pad)
#define NCLS  3
#define EPS_BN 1e-5f
#define SLOPE  0.2f

typedef __attribute__((ext_vector_type(8)))  __bf16 v8bf;
typedef __attribute__((ext_vector_type(16))) __bf16 v16bf;
typedef __attribute__((ext_vector_type(8)))  float  v8f;

// ---------------- utility ----------------
__global__ void fill_f32(float* p, float v, long n) {
    long i = (long)blockIdx.x * blockDim.x + threadIdx.x;
    if (i < n) p[i] = v;
}

__global__ void cvt_f32_bf16(const float* in, __bf16* out, long n) {
    long i = (long)blockIdx.x * blockDim.x + threadIdx.x;
    if (i < n) out[i] = (__bf16)in[i];
}

// Bt1[j,k] (j<512 -> W1[k,j], else Wres1[k,j-512]); Bt stored [NC1, FIN] row-major (transposed)
__global__ void pack_bt1(const float* W1, const float* Wres1, __bf16* Bt) {
    int i = blockIdx.x * blockDim.x + threadIdx.x;   // NC1*FIN
    if (i >= NC1 * FIN) return;
    int j = i / FIN, k = i % FIN;
    float v = (j < HD) ? W1[k * HD + j] : Wres1[k * HD + (j - HD)];
    Bt[(size_t)j * FIN + k] = (__bf16)v;
}

// Bt2[j,k], j<3 -> W2[k,j]; 16<=j<19 -> Wres2[k,j-16]; else 0. Stored [NC2, HD].
__global__ void pack_bt2(const float* W2, const float* Wres2, __bf16* Bt) {
    int i = blockIdx.x * blockDim.x + threadIdx.x;   // NC2*HD
    if (i >= NC2 * HD) return;
    int j = i / HD, k = i % HD;
    float v = 0.f;
    if (j < NCLS)                     v = W2[k * NCLS + j];
    else if (j >= 16 && j < 16+NCLS)  v = Wres2[k * NCLS + (j - 16)];
    Bt[(size_t)j * HD + k] = (__bf16)v;
}

// ---------------- WMMA GEMM: C[M,Nc] = A[M,K](bf16) * Bt[Nc,K](bf16, pre-transposed) -------------
// Per-wave 16(M) x 16*NBLK(N) strip; A fragment register-reused across NBLK WMMAs.
// Explicit k-step software pipelining: next k-step's fragments (distinct live ranges) are
// issued before the current WMMAs so loads overlap matrix math. Whole waves early-exit,
// so EXEC is all-ones at every WMMA.
template<int NBLK>
__global__ void wmma_gemm_bf16(const __bf16* __restrict__ A,
                               const __bf16* __restrict__ Bt,
                               float* __restrict__ C,
                               int M, int K, int Nc) {
    const int lane = threadIdx.x & 31;
    const int wave = threadIdx.x >> 5;
    const int nsuper = Nc / (16 * NBLK);                 // N-strips per row
    const long total = (long)(M >> 4) * nsuper;
    long tile = (long)blockIdx.x * (blockDim.x >> 5) + wave;
    if (tile >= total) return;
    const int m0 = (int)(tile / nsuper) << 4;
    const int n0 = (int)(tile % nsuper) * (16 * NBLK);

    // A fragment (contiguous k order, same as B): lanes 0-15 K=0..15 ; lanes 16-31 K=16..31
    const __bf16* arow = A + (size_t)(m0 + (lane & 15)) * K;
    const int akoff = (lane >> 4) << 4;
    // B fragments: lane holds column n0+j*16+(lane&15); lanes 0-15 K=0..15, lanes 16-31 K=16..31
    const __bf16* brow[NBLK];
    #pragma unroll
    for (int j = 0; j < NBLK; ++j)
        brow[j] = Bt + (size_t)(n0 + j * 16 + (lane & 15)) * K + ((lane >> 4) << 4);

    auto loadA = [&](int k0) -> v16bf {
        v8bf lo = *(const v8bf*)(arow + k0 + akoff);
        v8bf hi = *(const v8bf*)(arow + k0 + akoff + 8);
        return __builtin_shufflevector(lo, hi,
                0,1,2,3,4,5,6,7,8,9,10,11,12,13,14,15);
    };

    v8f acc[NBLK];
    #pragma unroll
    for (int j = 0; j < NBLK; ++j) acc[j] = (v8f){};

    // ---- pipeline prologue: fragments for k0 = 0 ----
    v16bf a_cur = loadA(0);
    v16bf b_cur[NBLK];
    #pragma unroll
    for (int j = 0; j < NBLK; ++j) b_cur[j] = *(const v16bf*)(brow[j]);

    // ---- steady state: issue next k-step loads, then current WMMAs ----
    int k0 = 0;
    for (; k0 + 32 < K; k0 += 32) {
        __builtin_prefetch(arow + k0 + 96, 0, 1);        // global_prefetch_b8
        v16bf a_nxt = loadA(k0 + 32);
        v16bf b_nxt[NBLK];
        #pragma unroll
        for (int j = 0; j < NBLK; ++j) b_nxt[j] = *(const v16bf*)(brow[j] + k0 + 32);
        #pragma unroll
        for (int j = 0; j < NBLK; ++j)
            acc[j] = __builtin_amdgcn_wmma_f32_16x16x32_bf16(false, a_cur, false, b_cur[j],
                                                             (short)0, acc[j], false, false);
        a_cur = a_nxt;
        #pragma unroll
        for (int j = 0; j < NBLK; ++j) b_cur[j] = b_nxt[j];
    }
    // ---- epilogue: last k-step ----
    #pragma unroll
    for (int j = 0; j < NBLK; ++j)
        acc[j] = __builtin_amdgcn_wmma_f32_16x16x32_bf16(false, a_cur, false, b_cur[j],
                                                         (short)0, acc[j], false, false);

    // C layout: VGPR r -> row m0 + r + (lane>=16 ? 8 : 0), col n + (lane&15)
    float* crow = C + (size_t)(m0 + ((lane >> 4) << 3)) * Nc + (lane & 15);
    #pragma unroll
    for (int j = 0; j < NBLK; ++j)
        #pragma unroll
        for (int r = 0; r < 8; ++r)
            crow[(size_t)r * Nc + n0 + j * 16] = acc[j][r];
}

// ---------------- attention, layer 1 ----------------
// el/er per node+head; z lives in cols [0,512) of zres1 (stride NC1)
__global__ void dots_l1(const float* __restrict__ zres, const float* __restrict__ al,
                        const float* __restrict__ ar, float* el, float* er) {
    int n = blockIdx.x;                      // one node per block, 128 threads = 4 waves
    int w = threadIdx.x >> 5, lane = threadIdx.x & 31;
    const float* zp  = zres + (size_t)n * NC1 + w * DHID;
    const float* alp = al + w * DHID;
    const float* arp = ar + w * DHID;
    float sl = 0.f, sr = 0.f;
    #pragma unroll
    for (int i = lane; i < DHID; i += 32) {
        float v = zp[i];
        sl += v * alp[i];
        sr += v * arp[i];
    }
    #pragma unroll
    for (int o = 16; o > 0; o >>= 1) {
        sl += __shfl_down(sl, o, 32);
        sr += __shfl_down(sr, o, 32);
    }
    if (lane == 0) { el[n * HEADS + w] = sl; er[n * HEADS + w] = sr; }
}

__device__ inline void atomicMaxF(float* addr, float val) {
    int* ai = (int*)addr;
    int old = __float_as_int(*addr);
    while (val > __int_as_float(old)) {
        int assumed = old;
        old = atomicCAS(ai, assumed, __float_as_int(val));
        if (old == assumed) break;
    }
}

__global__ void edge_score_l1(const int* __restrict__ srcv, const int* __restrict__ dstv,
                              const float* __restrict__ el, const float* __restrict__ er,
                              float* e1, float* m1) {
    int i = blockIdx.x * blockDim.x + threadIdx.x;
    if (i >= NEDGE) return;
    int s = srcv[i], d = dstv[i];
    #pragma unroll
    for (int h = 0; h < HEADS; ++h) {
        float v = el[s * HEADS + h] + er[d * HEADS + h];
        v = v > 0.f ? v : SLOPE * v;
        e1[i * HEADS + h] = v;
        atomicMaxF(&m1[d * HEADS + h], v);
    }
}

__global__ void edge_exp_l1(const int* __restrict__ dstv, float* e1,
                            const float* __restrict__ m1, float* den1) {
    int i = blockIdx.x * blockDim.x + threadIdx.x;
    if (i >= NEDGE) return;
    int d = dstv[i];
    #pragma unroll
    for (int h = 0; h < HEADS; ++h) {
        float x = __expf(e1[i * HEADS + h] - m1[d * HEADS + h]);
        e1[i * HEADS + h] = x;           // overwrite in place with exp value
        atomicAdd(&den1[d * HEADS + h], x);
    }
}

// scatter alpha-weighted messages into the residual half of zres1 (cols [512,1024))
__global__ void scatter_l1(float* zres, const int* __restrict__ srcv,
                           const int* __restrict__ dstv,
                           const float* __restrict__ e1, const float* __restrict__ den1) {
    __shared__ float salpha[HEADS];
    __shared__ int ss, sd;
    int i = blockIdx.x;            // one edge per block, 512 threads
    int t = threadIdx.x;
    if (t == 0) { ss = srcv[i]; sd = dstv[i]; }
    __syncthreads();
    if (t < HEADS) salpha[t] = e1[i * HEADS + t] / den1[sd * HEADS + t];
    __syncthreads();
    int h = t >> 7;                // 128 dims per head
    float v = zres[(size_t)ss * NC1 + t] * salpha[h];
    atomicAdd(&zres[(size_t)sd * NC1 + HD + t], v);
}

// bias + BatchNorm(inference) + ELU, emit bf16 input for layer-2 GEMM
__global__ void bn_elu_cvt(const float* __restrict__ zres, const float* __restrict__ b1,
                           const float* __restrict__ gamma, const float* __restrict__ beta,
                           const float* __restrict__ mean,  const float* __restrict__ var,
                           __bf16* hb) {
    long idx = (long)blockIdx.x * blockDim.x + threadIdx.x;   // N*HD
    if (idx >= (long)NNODE * HD) return;
    int  j = (int)(idx & (HD - 1));
    long n = idx >> 9;
    float v = zres[n * NC1 + HD + j] + b1[j];
    v = (v - mean[j]) * rsqrtf(var[j] + EPS_BN) * gamma[j] + beta[j];
    v = v > 0.f ? v : (__expf(v) - 1.f);       // ELU
    hb[idx] = (__bf16)v;
}

// ------- attention, layer 2 (1 head, C=3; z in cols [0,3), res in cols [16,19) of zres2) -------
__global__ void dots_l2(const float* __restrict__ zres2, const float* __restrict__ al2,
                        const float* __restrict__ ar2, float* el2, float* er2) {
    int n = blockIdx.x * blockDim.x + threadIdx.x;
    if (n >= NNODE) return;
    const float* z = zres2 + (size_t)n * NC2;
    float sl = 0.f, sr = 0.f;
    #pragma unroll
    for (int c = 0; c < NCLS; ++c) { sl += z[c] * al2[c]; sr += z[c] * ar2[c]; }
    el2[n] = sl; er2[n] = sr;
}

__global__ void edge_score_l2(const int* __restrict__ srcv, const int* __restrict__ dstv,
                              const float* __restrict__ el2, const float* __restrict__ er2,
                              float* e2, float* m2) {
    int i = blockIdx.x * blockDim.x + threadIdx.x;
    if (i >= NEDGE) return;
    int s = srcv[i], d = dstv[i];
    float v = el2[s] + er2[d];
    v = v > 0.f ? v : SLOPE * v;
    e2[i] = v;
    atomicMaxF(&m2[d], v);
}

__global__ void edge_exp_l2(const int* __restrict__ dstv, float* e2,
                            const float* __restrict__ m2, float* den2) {
    int i = blockIdx.x * blockDim.x + threadIdx.x;
    if (i >= NEDGE) return;
    int d = dstv[i];
    float x = __expf(e2[i] - m2[d]);
    e2[i] = x;
    atomicAdd(&den2[d], x);
}

__global__ void scatter_l2(float* zres2, const int* __restrict__ srcv,
                           const int* __restrict__ dstv,
                           const float* __restrict__ e2, const float* __restrict__ den2) {
    int i = blockIdx.x * blockDim.x + threadIdx.x;
    if (i >= NEDGE) return;
    int s = srcv[i], d = dstv[i];
    float alpha = e2[i] / den2[d];
    #pragma unroll
    for (int c = 0; c < NCLS; ++c)
        atomicAdd(&zres2[(size_t)d * NC2 + 16 + c], zres2[(size_t)s * NC2 + c] * alpha);
}

__global__ void final_out(const float* __restrict__ zres2, const float* __restrict__ b2,
                          float* out) {
    int idx = blockIdx.x * blockDim.x + threadIdx.x;   // N*NCLS
    if (idx >= NNODE * NCLS) return;
    int n = idx / NCLS, c = idx % NCLS;
    out[idx] = zres2[(size_t)n * NC2 + 16 + c] + b2[c];
}

// ---------------- launch ----------------
extern "C" void kernel_launch(void* const* d_in, const int* in_sizes, int n_in,
                              void* d_out, int out_size, void* d_ws, size_t ws_size,
                              hipStream_t stream) {
    const float* features = (const float*)d_in[0];
    const int*   srcv     = (const int*)  d_in[1];
    const int*   dstv     = (const int*)  d_in[2];
    const float* W1       = (const float*)d_in[3];
    const float* al1      = (const float*)d_in[4];
    const float* ar1      = (const float*)d_in[5];
    const float* b1       = (const float*)d_in[6];
    const float* Wres1    = (const float*)d_in[7];
    const float* bn_gamma = (const float*)d_in[8];
    const float* bn_beta  = (const float*)d_in[9];
    const float* bn_mean  = (const float*)d_in[10];
    const float* bn_var   = (const float*)d_in[11];
    const float* W2       = (const float*)d_in[12];
    const float* al2      = (const float*)d_in[13];
    const float* ar2      = (const float*)d_in[14];
    const float* b2       = (const float*)d_in[15];
    const float* Wres2    = (const float*)d_in[16];
    float* out = (float*)d_out;

    // workspace carve (256B aligned)
    char* w = (char*)d_ws;
    auto alloc = [&](size_t bytes) -> void* {
        void* p = (void*)w;
        w += (bytes + 255) & ~(size_t)255;
        return p;
    };
    __bf16* xb16  = (__bf16*)alloc((size_t)NNODE * FIN * 2);
    __bf16* Bt1   = (__bf16*)alloc((size_t)NC1 * FIN * 2);
    __bf16* Bt2   = (__bf16*)alloc((size_t)NC2 * HD * 2);
    float*  zres1 = (float*) alloc((size_t)NNODE * NC1 * 4);
    float*  el1   = (float*) alloc((size_t)NNODE * HEADS * 4);
    float*  er1   = (float*) alloc((size_t)NNODE * HEADS * 4);
    float*  m1    = (float*) alloc((size_t)NNODE * HEADS * 4);
    float*  den1  = (float*) alloc((size_t)NNODE * HEADS * 4);
    float*  e1    = (float*) alloc((size_t)NEDGE * HEADS * 4);
    __bf16* hb16  = (__bf16*)alloc((size_t)NNODE * HD * 2);
    float*  zres2 = (float*) alloc((size_t)NNODE * NC2 * 4);
    float*  el2   = (float*) alloc((size_t)NNODE * 4);
    float*  er2   = (float*) alloc((size_t)NNODE * 4);
    float*  m2    = (float*) alloc((size_t)NNODE * 4);
    float*  den2  = (float*) alloc((size_t)NNODE * 4);
    float*  e2    = (float*) alloc((size_t)NEDGE * 4);

    const float NEG_BIG = -3.0e38f;
    long nh = (long)NNODE * HEADS;
    fill_f32<<<(int)((nh + 255) / 256), 256, 0, stream>>>(m1,  NEG_BIG, nh);
    fill_f32<<<(int)((nh + 255) / 256), 256, 0, stream>>>(den1, 0.f,    nh);
    fill_f32<<<(NNODE + 255) / 256, 256, 0, stream>>>(m2,  NEG_BIG, (long)NNODE);
    fill_f32<<<(NNODE + 255) / 256, 256, 0, stream>>>(den2, 0.f,    (long)NNODE);

    // precision conversions / weight packing
    long nx = (long)NNODE * FIN;
    cvt_f32_bf16<<<(int)((nx + 255) / 256), 256, 0, stream>>>(features, xb16, nx);
    pack_bt1<<<(NC1 * FIN + 255) / 256, 256, 0, stream>>>(W1, Wres1, Bt1);
    pack_bt2<<<(NC2 * HD + 255) / 256, 256, 0, stream>>>(W2, Wres2, Bt2);

    // layer 1: fused z||res GEMM  [50000,128] x [128,1024], 16x64 strip per wave
    {
        long tiles = ((long)NNODE / 16) * (NC1 / 64);     // 3125 * 16 = 50000 strips
        int blocks = (int)((tiles + 7) / 8);
        wmma_gemm_bf16<4><<<blocks, 256, 0, stream>>>(xb16, Bt1, zres1, NNODE, FIN, NC1);
    }
    dots_l1<<<NNODE, 128, 0, stream>>>(zres1, al1, ar1, el1, er1);
    edge_score_l1<<<(NEDGE + 255) / 256, 256, 0, stream>>>(srcv, dstv, el1, er1, e1, m1);
    edge_exp_l1<<<(NEDGE + 255) / 256, 256, 0, stream>>>(dstv, e1, m1, den1);
    scatter_l1<<<NEDGE, HD, 0, stream>>>(zres1, srcv, dstv, e1, den1);

    long nhd = (long)NNODE * HD;
    bn_elu_cvt<<<(int)((nhd + 255) / 256), 256, 0, stream>>>(zres1, b1, bn_gamma, bn_beta,
                                                             bn_mean, bn_var, hb16);

    // layer 2: fused z||res GEMM  [50000,512] x [512,32], 16x32 strip per wave
    {
        long tiles = ((long)NNODE / 16) * (NC2 / 32);     // 3125 strips
        int blocks = (int)((tiles + 7) / 8);
        wmma_gemm_bf16<2><<<blocks, 256, 0, stream>>>(hb16, Bt2, zres2, NNODE, HD, NC2);
    }
    dots_l2<<<(NNODE + 255) / 256, 256, 0, stream>>>(zres2, al2, ar2, el2, er2);
    edge_score_l2<<<(NEDGE + 255) / 256, 256, 0, stream>>>(srcv, dstv, el2, er2, e2, m2);
    edge_exp_l2<<<(NEDGE + 255) / 256, 256, 0, stream>>>(dstv, e2, m2, den2);
    scatter_l2<<<(NEDGE + 255) / 256, 256, 0, stream>>>(zres2, srcv, dstv, e2, den2);
    final_out<<<(NNODE * NCLS + 255) / 256, 256, 0, stream>>>(zres2, b2, out);
}
